// MLA_60026462929155
// MI455X (gfx1250) — matrix operand order repaired and checked by reading, not compiled, not run on hardware
//
#include <hip/hip_runtime.h>

// ---------------------------------------------------------------------------
// MLA forward for MI455X (gfx1250, wave32, WMMA + TDM).
// All matmuls (5 GEMMs + flash attention) run on v_wmma_f32_16x16x32_bf16.
// Flash K/V tiles are staged into LDS by the Tensor Data Mover.
// GEMM: 128x64 block tile, double-buffered LDS, 4 WMMA / wave / K-step.
// ---------------------------------------------------------------------------

#ifndef MLA_USE_TDM
#define MLA_USE_TDM 1
#endif

typedef __bf16 bf16_t;
typedef __attribute__((ext_vector_type(16))) __bf16 v16bf;
typedef __attribute__((ext_vector_type(8)))  float   v8f;
typedef __attribute__((ext_vector_type(4)))  unsigned int u32x4;
typedef __attribute__((ext_vector_type(8)))  int i32x8;

static constexpr int kB   = 2;
static constexpr int kS   = 2048;
static constexpr int kDim = 2048;
static constexpr int kH   = 16;
static constexpr int kQL  = 512;    // q lora
static constexpr int kKVL = 512;    // kv lora
static constexpr int kDR  = 64;     // rope dims
static constexpr int kDV  = 128;    // value dims
static constexpr int kDQK = 192;    // 128 nope + 64 rope
static constexpr int kM   = kB * kS;         // 4096 token rows
#define SCALE_QK 0.07216878364870323f        // 192^-0.5

#define WMMA_BF16(a, b, c) \
  __builtin_amdgcn_wmma_f32_16x16x32_bf16(false, (a), false, (b), (short)0, (c), false, false)

// A-fragment (16x32 bf16, K-contiguous per lane): 16 consecutive bf16 @ p -> 2x b128
static __device__ __forceinline__ v16bf ld_a_frag(const bf16_t* p) {
  v16bf r;
  *(uint4*)&r       = *(const uint4*)p;
  *((uint4*)&r + 1) = *(const uint4*)(p + 8);
  return r;
}
// B-fragment when K-contiguous per lane: 16 consecutive bf16 -> 2x b128
static __device__ __forceinline__ v16bf ld_b_frag(const bf16_t* p) {
  v16bf r;
  *(uint4*)&r       = *(const uint4*)p;
  *((uint4*)&r + 1) = *(const uint4*)(p + 8);
  return r;
}

#if MLA_USE_TDM
// Issue a 2D TDM load (bf16 elements) global -> LDS with 4-dword padding per
// 32 dwords stored (row stride in LDS = data + data/8, keeps 16B alignment
// and spreads LDS banks). Wave-uniform args; one call per wave issues one DMA.
static __device__ __forceinline__ void tdm_load_2d(
    unsigned lds_off, const void* gaddr, unsigned tile_d0, unsigned tile_d1,
    unsigned tensor_d0, unsigned tensor_d1, unsigned long long d0_stride) {
  const unsigned long long ga = (unsigned long long)gaddr;
  u32x4 g0;
  g0[0] = 1u;                                   // count=1 (valid user D#)
  g0[1] = lds_off;                              // LDS byte address
  g0[2] = (unsigned)ga;                         // global_addr[31:0]
  g0[3] = (unsigned)((ga >> 32) & 0x1FFFFFFu)   // global_addr[56:32]
        | (2u << 30);                           // type = 2 ("image")
  i32x8 g1;
  g1[0] = (int)((1u << 16)      // data_size = 1 (2 bytes)
        | (1u << 20)            // pad_enable
        | (4u << 22)            // pad_interval: code 4 = 32 dwords
        | (3u << 25));          // pad_amount:   code 3 = 4 dwords
  g1[1] = (int)((tensor_d0 & 0xFFFFu) << 16);                  // dim0[15:0]
  g1[2] = (int)((tensor_d0 >> 16) | ((tensor_d1 & 0xFFFFu) << 16));
  g1[3] = (int)((tensor_d1 >> 16) | (tile_d0 << 16));
  g1[4] = (int)(tile_d1 & 0xFFFFu);                            // tile_dim1
  g1[5] = (int)(unsigned)(d0_stride & 0xFFFFFFFFu);
  g1[6] = (int)(unsigned)((d0_stride >> 32) & 0xFFFFu);
  g1[7] = 0;
  asm volatile("tensor_load_to_lds %0, %1" :: "s"(g0), "s"(g1) : "memory");
}
static __device__ __forceinline__ unsigned lds_byte_off(const void* p) {
  return (unsigned)(unsigned long long)
      (const __attribute__((address_space(3))) char*)p;
}
#endif

// ---------------------------------------------------------------------------
// f32 -> bf16 elementwise convert (grid-stride)
// ---------------------------------------------------------------------------
__global__ __launch_bounds__(256) void mla_cvt_bf16(const float* __restrict__ in,
                                                    bf16_t* __restrict__ out, long n) {
  long i  = (long)blockIdx.x * 256 + threadIdx.x;
  long st = (long)gridDim.x * 256;
  for (; i < n; i += st) out[i] = (bf16_t)in[i];
}

// f32 (K x N) -> bf16 transposed (N x K); K is a power of two (kshift = log2 K)
__global__ __launch_bounds__(256) void mla_cvt_bf16_t(const float* __restrict__ in,
                                                      bf16_t* __restrict__ out,
                                                      int kshift, int N, long total) {
  long i  = (long)blockIdx.x * 256 + threadIdx.x;
  long st = (long)gridDim.x * 256;
  const long kmask = (1L << kshift) - 1;
  for (; i < total; i += st) {
    const long n = i >> kshift, k = i & kmask;
    out[i] = (bf16_t)in[k * (long)N + n];
  }
}

// ---------------------------------------------------------------------------
// Tiled bf16 WMMA GEMM:  C[M,N] = A[M,K] * Bt[N,K]^T (+ bias[N]), f32 out.
// 256 threads = 8 waves; 128x64 block tile; BK=32; double-buffered LDS
// (single barrier per K-step, global prefetch overlaps WMMA).
// Wave w: M-tiles (w&3)*32+{0,16}, N-tiles (w>>2)*32+{0,16} -> 4 WMMA/step,
// each fragment reused twice.  B pre-transposed (N-major, K-contiguous).
// ---------------------------------------------------------------------------
__global__ __launch_bounds__(256) void mla_gemm_bf16(
    const bf16_t* __restrict__ A, const bf16_t* __restrict__ Bt,
    const float* __restrict__ bias, float* __restrict__ C, int N, int K) {
  __shared__ bf16_t As[2][128 * 40];   // 128 rows (m) x 32 (k), stride 40
  __shared__ bf16_t Bs[2][64 * 40];    // 64 rows (n) x 32 (k), stride 40
  const int tid  = threadIdx.x;
  const int lane = tid & 31;
  const int w    = tid >> 5;
  const int m0   = blockIdx.y * 128;
  const int n0   = blockIdx.x * 64;
  const int mw   = (w & 3) * 32;
  const int nw   = (w >> 2) * 32;
  const int am   = lane & 15;
  const int akb  = (lane >> 4) * 16;   // A frag K base
  const int bn   = lane & 15;
  const int bkb  = (lane >> 4) * 16;   // B frag K base
  const int srow = tid >> 2, sch = (tid & 3) * 8;  // staging: 4 chunks/row

  uint4 ra0, ra1, rb;
  auto gload = [&](int kt) {
    ra0 = *(const uint4*)&A[(size_t)(m0 + srow) * K + kt + sch];
    ra1 = *(const uint4*)&A[(size_t)(m0 + 64 + srow) * K + kt + sch];
    rb  = *(const uint4*)&Bt[(size_t)(n0 + srow) * K + kt + sch];
  };
  auto lstore = [&](int buf) {
    *(uint4*)&As[buf][srow * 40 + sch]        = ra0;
    *(uint4*)&As[buf][(64 + srow) * 40 + sch] = ra1;
    *(uint4*)&Bs[buf][srow * 40 + sch]        = rb;
  };

  v8f acc00 = {}, acc01 = {}, acc10 = {}, acc11 = {};
  gload(0);
  lstore(0);
  __syncthreads();
  const int nsteps = K >> 5;
  for (int s = 0; s < nsteps; ++s) {
    const int cur = s & 1;
    if (s + 1 < nsteps) gload((s + 1) * 32);           // prefetch next tile
    const v16bf af0 = ld_a_frag(&As[cur][(mw + am) * 40 + akb]);
    const v16bf af1 = ld_a_frag(&As[cur][(mw + 16 + am) * 40 + akb]);
    const v16bf bf0 = ld_b_frag(&Bs[cur][(nw + bn) * 40 + bkb]);
    const v16bf bf1 = ld_b_frag(&Bs[cur][(nw + 16 + bn) * 40 + bkb]);
    acc00 = WMMA_BF16(af0, bf0, acc00);
    acc01 = WMMA_BF16(af0, bf1, acc01);
    acc10 = WMMA_BF16(af1, bf0, acc10);
    acc11 = WMMA_BF16(af1, bf1, acc11);
    if (s + 1 < nsteps) lstore(cur ^ 1);               // fill other buffer
    __syncthreads();
  }

  const int row0 = m0 + mw + (lane >> 4) * 8;  // C/D layout: row r + 8*half
  const int col0 = n0 + nw + (lane & 15);
  float bz0 = 0.f, bz1 = 0.f;
  if (bias) { bz0 = bias[col0]; bz1 = bias[col0 + 16]; }
#pragma unroll
  for (int r = 0; r < 8; ++r) {
    C[(size_t)(row0 + r) * N + col0]           = acc00[r] + bz0;
    C[(size_t)(row0 + r) * N + col0 + 16]      = acc01[r] + bz1;
    C[(size_t)(row0 + 16 + r) * N + col0]      = acc10[r] + bz0;
    C[(size_t)(row0 + 16 + r) * N + col0 + 16] = acc11[r] + bz1;
  }
}

// ---------------------------------------------------------------------------
// RMSNorm: one block per row; normalize first W columns of a row of width
// inStride; writes bf16 rows of width W (feeds next GEMM).
// ---------------------------------------------------------------------------
__global__ __launch_bounds__(256) void mla_rmsnorm_bf16(
    const float* __restrict__ x, const float* __restrict__ wgt,
    bf16_t* __restrict__ out, int inStride, int W) {
  __shared__ float red[256];
  const int row = blockIdx.x;
  const float* xr = x + (size_t)row * inStride;
  float ss = 0.f;
  for (int i = threadIdx.x; i < W; i += 256) { float v = xr[i]; ss += v * v; }
  red[threadIdx.x] = ss;
  __syncthreads();
  for (int s = 128; s > 0; s >>= 1) {
    if (threadIdx.x < s) red[threadIdx.x] += red[threadIdx.x + s];
    __syncthreads();
  }
  const float inv = rsqrtf(red[0] / (float)W + 1e-6f);
  for (int i = threadIdx.x; i < W; i += 256)
    out[(size_t)row * W + i] = (bf16_t)(xr[i] * inv * wgt[i]);
}

// ---------------------------------------------------------------------------
// Q pack: q f32 (B*S, H*192) -> bf16 [B][H][S][192], RoPE on dims 128..191.
// 192 threads: d = tid, loop over heads (no integer divides).
// ---------------------------------------------------------------------------
__global__ __launch_bounds__(192) void mla_q_pack(
    const float* __restrict__ q, const float* __restrict__ fcos,
    const float* __restrict__ fsin, bf16_t* __restrict__ qb) {
  const int row = blockIdx.x;               // b*S + s
  const int s = row & (kS - 1), bb = row >> 11;
  const int d = threadIdx.x;                // 0..191
  const float* qr = q + (size_t)row * (kH * kDQK);
  const bool pe = d >= 128;
  int i = 0; float c = 0.f, sn = 0.f;
  if (pe) { i = (d - 128) >> 1; c = fcos[s * 32 + i]; sn = fsin[s * 32 + i]; }
  bf16_t* dst = qb + ((size_t)bb * kH * kS + s) * kDQK + d;
#pragma unroll 4
  for (int h = 0; h < kH; ++h) {
    float val;
    if (!pe) {
      val = qr[h * kDQK + d];
    } else {
      const float x0 = qr[h * kDQK + 128 + 2 * i];
      const float x1 = qr[h * kDQK + 128 + 2 * i + 1];
      val = ((d & 1) == 0) ? (x0 * c - x1 * sn) : (x0 * sn + x1 * c);
    }
    dst[(size_t)h * kS * kDQK] = (bf16_t)val;
  }
}

// ---------------------------------------------------------------------------
// K pack: kvb f32 (B*S, H*256) nope part + k_pe from kv_full cols 512..575
// (RoPE, broadcast over heads) -> K bf16 [B][H][S][192].
// ---------------------------------------------------------------------------
__global__ __launch_bounds__(192) void mla_k_pack(
    const float* __restrict__ kvb, const float* __restrict__ kvfull,
    const float* __restrict__ fcos, const float* __restrict__ fsin,
    bf16_t* __restrict__ ko) {
  const int row = blockIdx.x;
  const int s = row & (kS - 1), bb = row >> 11;
  const int d = threadIdx.x;                // 0..191
  const float* kr = kvb + (size_t)row * (kH * 256);
  const float* pe = kvfull + (size_t)row * (kKVL + kDR) + kKVL;
  const bool ispe = d >= 128;
  float peval = 0.f;
  if (ispe) {   // rope'd k_pe: identical across heads, compute once
    const int i = (d - 128) >> 1;
    const float c = fcos[s * 32 + i], sn = fsin[s * 32 + i];
    const float x0 = pe[2 * i], x1 = pe[2 * i + 1];
    peval = ((d & 1) == 0) ? (x0 * c - x1 * sn) : (x0 * sn + x1 * c);
  }
  bf16_t* dst = ko + ((size_t)bb * kH * kS + s) * kDQK + d;
#pragma unroll 4
  for (int h = 0; h < kH; ++h) {
    const float val = ispe ? peval : kr[h * 256 + d];
    dst[(size_t)h * kS * kDQK] = (bf16_t)val;
  }
}

// ---------------------------------------------------------------------------
// V pack (transposed): kvb f32 -> V bf16 [B][H][dv=128][S].  One block per
// (b,h,dv) row; writes coalesced along s.  Transposed V makes the P*V
// B-fragments K-contiguous in LDS (b128 loads instead of 16 scalar loads).
// ---------------------------------------------------------------------------
__global__ __launch_bounds__(256) void mla_v_pack_t(const float* __restrict__ kvb,
                                                    bf16_t* __restrict__ vt) {
  const int bhdv = blockIdx.x;          // 0 .. kB*kH*kDV-1
  const int dv = bhdv & 127;
  const int bh = bhdv >> 7;
  const int h  = bh & 15, bb = bh >> 4;
  bf16_t* dst = vt + (size_t)bhdv * kS;
  const float* src = kvb + (size_t)bb * kS * (kH * 256) + h * 256 + 128 + dv;
  for (int s = threadIdx.x; s < kS; s += 256)
    dst[s] = (bf16_t)src[(size_t)s * (kH * 256)];
}

// ---------------------------------------------------------------------------
// Flash attention (causal). Block = 128 threads = 4 waves; 64 q rows/block
// (16 per wave), 64-key tiles. K (64x192) and V^T (128x64) tiles staged into
// LDS by the Tensor Data Mover (pad: 4 dwords per 32 -> row strides 216/72
// elements, 16B-aligned, bank-friendly). Online softmax via half-wave
// shfl_xor reductions; P round-trips per-wave LDS (C-layout -> A-layout).
// ---------------------------------------------------------------------------
__global__ __launch_bounds__(128) void mla_flash(
    const bf16_t* __restrict__ Q, const bf16_t* __restrict__ Kt,
    const bf16_t* __restrict__ Vt, bf16_t* __restrict__ Ob) {
  __shared__ bf16_t Ks[64 * 216];      // 64 keys x 192 d, stride 216 (432B)
  __shared__ bf16_t Vs[128 * 72];      // 128 dv x 64 keys, stride 72 (144B)
  __shared__ bf16_t Ps[4 * 16 * 72];   // per-wave 16x64 P, stride 72
  const int tid = threadIdx.x, lane = tid & 31, w = tid >> 5;
  const int qblk = blockIdx.x, h = blockIdx.y, bb = blockIdx.z;
  const int qbase = qblk * 64;
  const size_t bh = (size_t)bb * kH + h;
  const bf16_t* Qg = Q + (bh * kS + qbase + w * 16) * kDQK;
  const bf16_t* Kg = Kt + bh * (size_t)kS * kDQK;
  const bf16_t* Vg = Vt + bh * (size_t)kDV * kS;   // transposed [dv][s]
  const int am = lane & 15, akb = (lane >> 4) * 16;
  const int bn = lane & 15, bkb = (lane >> 4) * 16;
  const int hl = lane >> 4;

  // Q fragments (6 chunks of 32 along d), vector loads from global.
  v16bf qf[6];
#pragma unroll
  for (int kc = 0; kc < 6; ++kc)
    qf[kc] = ld_a_frag(&Qg[(size_t)am * kDQK + kc * 32 + akb]);

  v8f acc[8];
#pragma unroll
  for (int dt = 0; dt < 8; ++dt) acc[dt] = (v8f){};
  float mrow[8], lrow[8];
#pragma unroll
  for (int r = 0; r < 8; ++r) { mrow[r] = -1e30f; lrow[r] = 0.f; }

  const int nkb = qblk + 1;   // causal: only key blocks with k0 <= q_max
  for (int kb = 0; kb < nkb; ++kb) {
    const int k0 = kb * 64;
#if MLA_USE_TDM
    if (tid < 32) {           // wave 0 issues both tile DMAs
      tdm_load_2d(lds_byte_off(Ks), Kg + (size_t)k0 * kDQK,
                  /*tile*/ kDQK, 64, /*tensor*/ kDQK, 64, /*stride*/ kDQK);
      tdm_load_2d(lds_byte_off(Vs), Vg + k0,
                  /*tile*/ 64, kDV, /*tensor*/ 64, kDV, /*stride*/ kS);
#if __has_builtin(__builtin_amdgcn_s_wait_tensorcnt)
      __builtin_amdgcn_s_wait_tensorcnt(0);
#else
      asm volatile("s_wait_tensorcnt 0x0" ::: "memory");
#endif
    }
    __syncthreads();
#else
    for (int i = tid; i < 64 * 24; i += 128) {       // K tile, padded layout
      const int r = i / 24, e = (i - r * 24) * 8;
      *(uint4*)&Ks[r * 216 + e + ((e >> 6) << 3)] =
          *(const uint4*)&Kg[(size_t)(k0 + r) * kDQK + e];
    }
    for (int i = tid; i < 128 * 8; i += 128) {       // V^T tile
      const int r = i >> 3, e = (i & 7) * 8;
      *(uint4*)&Vs[r * 72 + e] = *(const uint4*)&Vg[(size_t)r * kS + k0 + e];
    }
    __syncthreads();
#endif

    // scores: S = Q(16x192) * K^T(192x64) -> 4 N-tiles
    v8f sc[4];
#pragma unroll
    for (int nt = 0; nt < 4; ++nt) {
      v8f cf = {};
#pragma unroll
      for (int kc = 0; kc < 6; ++kc) {
        const int e = kc * 32 + bkb;
        const v16bf bt = ld_b_frag(&Ks[(nt * 16 + bn) * 216 + e + ((e >> 6) << 3)]);
        cf = WMMA_BF16(qf[kc], bt, cf);
      }
      sc[nt] = cf;
    }

    // scale + causal mask + online softmax, write P (bf16) to per-wave LDS
#pragma unroll
    for (int r = 0; r < 8; ++r) {
      const int qrow = qbase + w * 16 + r + hl * 8;
      float sv[4];
      float mx = -1e30f;
#pragma unroll
      for (int nt = 0; nt < 4; ++nt) {
        const int key = k0 + nt * 16 + (lane & 15);
        float v = sc[nt][r] * SCALE_QK;
        if (key > qrow) v = -1e30f;
        sv[nt] = v;
        mx = fmaxf(mx, v);
      }
#pragma unroll
      for (int off = 1; off < 16; off <<= 1) mx = fmaxf(mx, __shfl_xor(mx, off, 32));
      const float mnew = fmaxf(mrow[r], mx);
      const float alpha = __expf(mrow[r] - mnew);
      float rs = 0.f;
#pragma unroll
      for (int nt = 0; nt < 4; ++nt) {
        const float p = __expf(sv[nt] - mnew);
        rs += p;
        Ps[w * 16 * 72 + (r + hl * 8) * 72 + nt * 16 + (lane & 15)] = (bf16_t)p;
      }
#pragma unroll
      for (int off = 1; off < 16; off <<= 1) rs += __shfl_xor(rs, off, 32);
      lrow[r] = lrow[r] * alpha + rs;
      mrow[r] = mnew;
#pragma unroll
      for (int dt = 0; dt < 8; ++dt) acc[dt][r] = acc[dt][r] * alpha;
    }
    asm volatile("s_wait_dscnt 0" ::: "memory");  // wave-local P write->read

    // O += P(16x64) * V(64x128): 2 key-chunks x 8 dv-tiles
#pragma unroll
    for (int kc2 = 0; kc2 < 2; ++kc2) {
      const v16bf a = ld_a_frag(&Ps[w * 16 * 72 + am * 72 + kc2 * 32 + akb]);
#pragma unroll
      for (int dt = 0; dt < 8; ++dt) {
        const v16bf bv = ld_b_frag(&Vs[(dt * 16 + bn) * 72 + kc2 * 32 + bkb]);
        acc[dt] = WMMA_BF16(a, bv, acc[dt]);
      }
    }
    __syncthreads();  // protect K/V tiles before next iteration reload
  }

  // epilogue: O /= l, write bf16 (B*S, H*128)
#pragma unroll
  for (int r = 0; r < 8; ++r) {
    const int srow = qbase + w * 16 + r + hl * 8;
    const float inv = 1.f / lrow[r];
    const size_t orow = ((size_t)bb * kS + srow) * (kH * kDV) + h * kDV;
#pragma unroll
    for (int dt = 0; dt < 8; ++dt)
      Ob[orow + dt * 16 + (lane & 15)] = (bf16_t)(acc[dt][r] * inv);
  }
}

// ---------------------------------------------------------------------------
// Host-side orchestration
// ---------------------------------------------------------------------------
extern "C" void kernel_launch(void* const* d_in, const int* in_sizes, int n_in,
                              void* d_out, int out_size, void* d_ws, size_t ws_size,
                              hipStream_t stream) {
  (void)in_sizes; (void)n_in; (void)out_size; (void)ws_size;
  const float* x       = (const float*)d_in[0];
  const float* wq_a    = (const float*)d_in[1];
  const float* wq_a_b  = (const float*)d_in[2];
  const float* q_norm  = (const float*)d_in[3];
  const float* wq_b    = (const float*)d_in[4];
  const float* wkv_a   = (const float*)d_in[5];
  const float* wkv_a_b = (const float*)d_in[6];
  const float* kv_norm = (const float*)d_in[7];
  const float* wkv_b   = (const float*)d_in[8];
  const float* wo      = (const float*)d_in[9];
  const float* fcos    = (const float*)d_in[10];
  const float* fsin    = (const float*)d_in[11];
  // d_in[12] = mask (causal implemented directly), d_in[13] = start_pos (0)

  char* ws = (char*)d_ws;
  size_t off = 0;
  auto carve = [&](size_t bytes) -> void* {
    void* p = ws + off;
    off = (off + bytes + 255) & ~(size_t)255;
    return p;
  };
  bf16_t* xb     = (bf16_t*)carve((size_t)kM * kDim * 2);
  bf16_t* wqa_t  = (bf16_t*)carve((size_t)kDim * kQL * 2);          // (512, 2048)
  bf16_t* wqb_t  = (bf16_t*)carve((size_t)kQL * kH * kDQK * 2);     // (3072, 512)
  bf16_t* wkva_t = (bf16_t*)carve((size_t)kDim * (kKVL + kDR) * 2); // (576, 2048)
  bf16_t* wkvb_t = (bf16_t*)carve((size_t)kKVL * kH * 256 * 2);     // (4096, 512)
  bf16_t* wo_t   = (bf16_t*)carve((size_t)(kH * kDV) * kDim * 2);   // (2048, 2048)
  float*  qa     = (float*)carve((size_t)kM * kQL * 4);
  bf16_t* qan    = (bf16_t*)carve((size_t)kM * kQL * 2);
  float*  qf32   = (float*)carve((size_t)kM * kH * kDQK * 4);
  float*  kvf    = (float*)carve((size_t)kM * (kKVL + kDR) * 4);
  bf16_t* kvn    = (bf16_t*)carve((size_t)kM * kKVL * 2);
  float*  kvb    = (float*)carve((size_t)kM * kH * 256 * 4);
  bf16_t* qb     = (bf16_t*)carve((size_t)kM * kH * kDQK * 2);
  bf16_t* kbuf   = (bf16_t*)carve((size_t)kM * kH * kDQK * 2);
  bf16_t* vbuf   = (bf16_t*)carve((size_t)kM * kH * kDV * 2);       // transposed
  bf16_t* attnb  = (bf16_t*)carve((size_t)kM * kH * kDV * 2);

  // 1. bf16 conversions: x as-is; all weights transposed to (N, K)
  mla_cvt_bf16<<<4096, 256, 0, stream>>>(x, xb, (long)kM * kDim);
  mla_cvt_bf16_t<<<1024, 256, 0, stream>>>(wq_a, wqa_t, 11, kQL, (long)kDim * kQL);
  mla_cvt_bf16_t<<<1024, 256, 0, stream>>>(wq_b, wqb_t, 9, kH * kDQK, (long)kQL * kH * kDQK);
  mla_cvt_bf16_t<<<1024, 256, 0, stream>>>(wkv_a, wkva_t, 11, kKVL + kDR, (long)kDim * (kKVL + kDR));
  mla_cvt_bf16_t<<<1024, 256, 0, stream>>>(wkv_b, wkvb_t, 9, kH * 256, (long)kKVL * kH * 256);
  mla_cvt_bf16_t<<<2048, 256, 0, stream>>>(wo, wo_t, 11, kDim, (long)(kH * kDV) * kDim);

  // 2. q_a = x @ wq_a + b      (4096x512, K=2048)
  mla_gemm_bf16<<<dim3(kQL / 64, kM / 128), 256, 0, stream>>>(xb, wqa_t, wq_a_b, qa, kQL, kDim);
  // 3. rmsnorm(q_a) -> bf16
  mla_rmsnorm_bf16<<<kM, 256, 0, stream>>>(qa, q_norm, qan, kQL, kQL);
  // 4. q = qan @ wq_b          (4096x3072, K=512)
  mla_gemm_bf16<<<dim3((kH * kDQK) / 64, kM / 128), 256, 0, stream>>>(qan, wqb_t, nullptr, qf32, kH * kDQK, kQL);
  // 5. kv_full = x @ wkv_a + b (4096x576, K=2048)
  mla_gemm_bf16<<<dim3((kKVL + kDR) / 64, kM / 128), 256, 0, stream>>>(xb, wkva_t, wkv_a_b, kvf, kKVL + kDR, kDim);
  // 6. rmsnorm(kv[:, :512]) -> bf16
  mla_rmsnorm_bf16<<<kM, 256, 0, stream>>>(kvf, kv_norm, kvn, kKVL + kDR, kKVL);
  // 7. kvb = kvn @ wkv_b       (4096x4096, K=512)
  mla_gemm_bf16<<<dim3((kH * 256) / 64, kM / 128), 256, 0, stream>>>(kvn, wkvb_t, nullptr, kvb, kH * 256, kKVL);
  // 8. RoPE + relayout (K row-major per head, V transposed per head)
  mla_q_pack<<<kM, 192, 0, stream>>>(qf32, fcos, fsin, qb);
  mla_k_pack<<<kM, 192, 0, stream>>>(kvb, kvf, fcos, fsin, kbuf);
  mla_v_pack_t<<<kB * kH * kDV, 256, 0, stream>>>(kvb, vbuf);
  // 9. causal flash attention (TDM-staged K/V tiles)
  mla_flash<<<dim3(kS / 64, kH, kB), 128, 0, stream>>>(qb, kbuf, vbuf, attnb);
  // 10. out = attn @ wo        (4096x2048, K=2048), f32 output
  mla_gemm_bf16<<<dim3(kDim / 64, kM / 128), 256, 0, stream>>>(attnb, wo_t, nullptr, (float*)d_out, kDim, kH * kDV);
}
